// RetinaNet_22746146799747
// MI455X (gfx1250) — matrix operand order repaired and checked by reading, not compiled, not run on hardware
//
#include <hip/hip_runtime.h>
#include <hip/hip_bf16.h>

// ---------------------------------------------------------------------------
// RetinaNet head on gfx1250: implicit-GEMM 3x3 convs via v_wmma_f32_16x16x32_bf16
// ---------------------------------------------------------------------------

typedef __bf16 v8bf  __attribute__((ext_vector_type(8)));
typedef __bf16 v16bf __attribute__((ext_vector_type(16)));
typedef float  v8f   __attribute__((ext_vector_type(8)));

__device__ __forceinline__ unsigned short f2bf(float f) {
    unsigned u = __builtin_bit_cast(unsigned, f);
    unsigned r = u + 0x7FFFu + ((u >> 16) & 1u);   // round-to-nearest-even
    return (unsigned short)(r >> 16);
}

__device__ __forceinline__ v16bf cat16(v8bf a, v8bf b) {
    return __builtin_shufflevector(a, b, 0,1,2,3,4,5,6,7,8,9,10,11,12,13,14,15);
}

// ---------------------------------------------------------------------------
// Weight packing: f32 [L][CO][CI][3][3]  ->  bf16 [L][CO][9][CI]
// ---------------------------------------------------------------------------
__global__ __launch_bounds__(256) void pack_tower_k(const float* __restrict__ src,
                                                    unsigned short* __restrict__ dst,
                                                    int total) {
    int idx = blockIdx.x * 256 + threadIdx.x;
    if (idx >= total) return;
    int ci = idx & 255;
    int t  = idx >> 8;
    int r  = t % 9;
    int t2 = t / 9;
    int co = t2 & 255;
    int l  = t2 >> 8;
    dst[idx] = f2bf(src[(((size_t)(l * 256 + co) * 256 + ci) * 9) + r]);
}

// f32 [CO][256][3][3] -> bf16 [CO_pad][9][256] (zero padded co)
__global__ __launch_bounds__(256) void pack_out_k(const float* __restrict__ src,
                                                  unsigned short* __restrict__ dst,
                                                  int CO, int total) {
    int idx = blockIdx.x * 256 + threadIdx.x;
    if (idx >= total) return;
    int ci = idx & 255;
    int t  = idx >> 8;
    int r  = t % 9;
    int co = t / 9;
    float v = (co < CO) ? src[((size_t)(co * 256 + ci) * 9) + r] : 0.0f;
    dst[idx] = f2bf(v);
}

// NCHW f32 -> NHWC bf16
__global__ __launch_bounds__(256) void nchw2nhwc_k(const float* __restrict__ src,
                                                   unsigned short* __restrict__ dst,
                                                   int HW) {
    int idx = blockIdx.x * 256 + threadIdx.x;
    int total = 4 * HW * 256;
    if (idx >= total) return;
    int c   = idx & 255;
    int pos = (idx >> 8) % HW;
    int b   = idx / (HW * 256);
    dst[idx] = f2bf(src[((size_t)(b * 256 + c)) * HW + pos]);
}

// ---------------------------------------------------------------------------
// Implicit-GEMM 3x3 SAME conv, stride 1, CI=256 fixed.
//   X : NHWC bf16 [B,H,W,256]
//   Wp: bf16 [CO_pad][9][256]
// mode 0: Y (NHWC bf16) = relu(conv + bias), CO=256
// mode 1: scatter f32 conv+bias into detection-head layout
//         Hout[b][lvl_base + (h*W+w)*A + co/K][co%K]
// Workgroup: 256 thr = 8 waves; tile 128co x 64pos; wave tile 32co x 32pos.
// ---------------------------------------------------------------------------
__global__ __launch_bounds__(256) void conv3x3_wmma_k(
    const unsigned short* __restrict__ X,
    const unsigned short* __restrict__ Wp,
    const float* __restrict__ bias,
    unsigned short* __restrict__ Y,
    float* __restrict__ Hout,
    int H, int W, int CO, int mode,
    int A, int K, int level_base, int Ntot)
{
    __shared__ alignas(16) unsigned short Xs[6400];  // (TH+2)*(W+2)*32 <= 6336

    const int HW   = H * W;
    const int TH   = 64 / W;          // rows per 64-position tile (W in {64,32,16,8})
    const int Wp2  = W + 2;
    const int sites = (TH + 2) * Wp2;

    const int tid  = threadIdx.x;
    const int lane = tid & 31;
    const int wv   = tid >> 5;

    const int co_tile = blockIdx.x;              // *128
    const int ptile   = blockIdx.y;              // *64
    const int tiles_per_img = HW / 64;
    const int b  = ptile / tiles_per_img;
    const int p0 = (ptile % tiles_per_img) * 64;
    const int h0 = p0 / W;

    const int wco  = (wv & 3) * 32;              // wave co offset within 128
    const int wpos = (wv >> 2) * 32;             // wave pos offset within 64
    const int ln   = lane & 15;
    const int hi   = lane >> 4;                  // lane group
    const int kbA  = hi * 8;                     // A frag K chunks: {kbA..kbA+7, kbA+16..}
    const int kbB  = hi * 8;                     // B frag K chunks: {kbB..kbB+7, kbB+16..}

    // per-lane B positions (independent of filter offset)
    int lr[2], lc[2];
#pragma unroll
    for (int s = 0; s < 2; ++s) {
        int pl = wpos + s * 16 + ln;
        lr[s] = pl / W;
        lc[s] = pl % W;
    }

    v8f acc[2][2];
#pragma unroll
    for (int i = 0; i < 2; ++i)
#pragma unroll
        for (int j = 0; j < 2; ++j)
            acc[i][j] = (v8f)(0.0f);

    for (int kb = 0; kb < 8; ++kb) {             // 8 blocks of 32 input channels
        __syncthreads();
        if (tid < sites) {
            int rr = tid / Wp2, cc = tid % Wp2;
            int gh = h0 - 1 + rr, gw = cc - 1;
            uint4* dst = (uint4*)&Xs[tid * 32];
            if (gh >= 0 && gh < H && gw >= 0 && gw < W) {
                const uint4* src =
                    (const uint4*)(X + (((size_t)(b * H + gh) * W + gw) * 256 + kb * 32));
                dst[0] = src[0]; dst[1] = src[1]; dst[2] = src[2]; dst[3] = src[3];
            } else {
                uint4 z = {0u, 0u, 0u, 0u};
                dst[0] = z; dst[1] = z; dst[2] = z; dst[3] = z;
            }
        }
        __syncthreads();

#pragma unroll
        for (int r = 0; r < 9; ++r) {
            const int kh = r / 3, kw = r % 3;

            v16bf afr[2];
#pragma unroll
            for (int s = 0; s < 2; ++s) {
                int co_g = co_tile * 128 + wco + s * 16 + ln;
                const unsigned short* ap =
                    Wp + ((size_t)co_g * 9 + r) * 256 + kb * 32 + kbA;
                v8bf a0 = *(const v8bf*)ap;
                v8bf a1 = *(const v8bf*)(ap + 16);
                afr[s] = cat16(a0, a1);
            }

            v16bf bfr[2];
#pragma unroll
            for (int s = 0; s < 2; ++s) {
                int pr = lr[s] + kh, pc = lc[s] + kw;
                const unsigned short* bp = &Xs[(pr * Wp2 + pc) * 32 + kbB];
                v8bf b0 = *(const v8bf*)bp;
                v8bf b1 = *(const v8bf*)(bp + 16);
                bfr[s] = cat16(b0, b1);
            }

#pragma unroll
            for (int ci = 0; ci < 2; ++ci)
#pragma unroll
                for (int pi = 0; pi < 2; ++pi)
                    acc[ci][pi] = __builtin_amdgcn_wmma_f32_16x16x32_bf16(
                        false, afr[ci], false, bfr[pi],
                        (short)0, acc[ci][pi], false, false);
        }
    }

    // C/D layout: VGPR j -> M = j + 8*hi ; N = lane & 15
#pragma unroll
    for (int ci = 0; ci < 2; ++ci) {
#pragma unroll
        for (int pi = 0; pi < 2; ++pi) {
#pragma unroll
            for (int j = 0; j < 8; ++j) {
                int m    = j + hi * 8;
                int co_g = co_tile * 128 + wco + ci * 16 + m;
                int pl   = wpos + pi * 16 + ln;
                int p    = p0 + pl;
                int h    = p / W, w = p % W;
                float v  = acc[ci][pi][j];
                if (mode == 0) {
                    v += bias[co_g];
                    v = v > 0.0f ? v : 0.0f;
                    Y[(((size_t)(b * H + h)) * W + w) * 256 + co_g] = f2bf(v);
                } else if (co_g < CO) {
                    v += bias[co_g];
                    int aidx = co_g / K;
                    int kk   = co_g - aidx * K;
                    size_t o = ((size_t)b * Ntot + level_base + (h * W + w) * A + aidx);
                    Hout[o * K + kk] = v;
                }
            }
        }
    }
}

// ---------------------------------------------------------------------------
// Anchors: torchvision AnchorGenerator + clip. N = 48960 per image, B = 4.
// ---------------------------------------------------------------------------
__global__ __launch_bounds__(256) void anchors_k(const int* __restrict__ ih,
                                                 const int* __restrict__ iw,
                                                 float* __restrict__ out) {
    int n = blockIdx.x * 256 + threadIdx.x;
    if (n >= 48960) return;
    const int   Hs[4]    = {64, 32, 16, 8};
    const int   base[4]  = {0, 36864, 46080, 48384};
    const float sc[4][3] = {{32.f, 40.f, 50.f},
                            {64.f, 80.f, 101.f},
                            {128.f, 161.f, 203.f},
                            {256.f, 322.f, 406.f}};
    int l = 3;
    if (n < 36864) l = 0; else if (n < 46080) l = 1; else if (n < 48384) l = 2;
    int nl   = n - base[l];
    int a    = nl % 9;
    int cell = nl / 9;
    int Wl   = Hs[l];
    int h = cell / Wl, w = cell % Wl;

    int IH = *ih, IW = *iw;
    int sh = IH / Hs[l], sw = IW / Hs[l];

    const float hr_t[3] = {0.70710678118654752f, 1.0f, 1.41421356237309515f};
    int   ri = a / 3, si = a % 3;
    float hr = hr_t[ri], wr = 1.0f / hr;
    float s  = sc[l][si];
    float ws = wr * s, hs = hr * s;
    float bx1 = rintf(-ws * 0.5f), by1 = rintf(-hs * 0.5f);
    float bx2 = rintf( ws * 0.5f), by2 = rintf( hs * 0.5f);
    float sx = (float)(w * sw), sy = (float)(h * sh);
    float fW = (float)IW, fH = (float)IH;
    float x1 = fminf(fmaxf(sx + bx1, 0.f), fW);
    float y1 = fminf(fmaxf(sy + by1, 0.f), fH);
    float x2 = fminf(fmaxf(sx + bx2, 0.f), fW);
    float y2 = fminf(fmaxf(sy + by2, 0.f), fH);
#pragma unroll
    for (int b = 0; b < 4; ++b) {
        float* o = out + ((size_t)b * 48960 + n) * 4;
        o[0] = x1; o[1] = y1; o[2] = x2; o[3] = y2;
    }
}

// ---------------------------------------------------------------------------
extern "C" void kernel_launch(void* const* d_in, const int* in_sizes, int n_in,
                              void* d_out, int out_size, void* d_ws, size_t ws_size,
                              hipStream_t stream) {
    (void)in_sizes; (void)n_in; (void)out_size; (void)ws_size;

    const float* feats[4] = {(const float*)d_in[0], (const float*)d_in[1],
                             (const float*)d_in[2], (const float*)d_in[3]};
    const float* cls_tw   = (const float*)d_in[4];
    const float* cls_tb   = (const float*)d_in[5];
    const float* cls_wout = (const float*)d_in[6];
    const float* cls_bout = (const float*)d_in[7];
    const float* reg_tw   = (const float*)d_in[8];
    const float* reg_tb   = (const float*)d_in[9];
    const float* reg_wout = (const float*)d_in[10];
    const float* reg_bout = (const float*)d_in[11];
    const int*   img_h    = (const int*)d_in[12];
    const int*   img_w    = (const int*)d_in[13];

    // workspace layout (bytes)
    char* ws = (char*)d_ws;
    unsigned short* Wp_cls_t = (unsigned short*)(ws + 0);           // 4*256*9*256 bf16
    unsigned short* Wp_reg_t = (unsigned short*)(ws + 4718592);
    unsigned short* Wp_cls_o = (unsigned short*)(ws + 9437184);     // 768*9*256
    unsigned short* Wp_reg_o = (unsigned short*)(ws + 12976128);    // 128*9*256
    unsigned short* Xbuf     = (unsigned short*)(ws + 13565952);    // 4*5440*256
    unsigned short* ping     = (unsigned short*)(ws + 24707072);    // 4*4096*256
    unsigned short* pong     = (unsigned short*)(ws + 33095680);

    const int Hs[4]       = {64, 32, 16, 8};
    const int HWs[4]      = {4096, 1024, 256, 64};
    const size_t xoff[4]  = {0, 4194304, 5242880, 5505024};         // elements
    const int lvl_base[4] = {0, 36864, 46080, 48384};
    const int NTOT        = 48960;

    // 1) pack weights to bf16 [co][r][ci]
    {
        int t = 4 * 256 * 9 * 256;
        pack_tower_k<<<(t + 255) / 256, 256, 0, stream>>>(cls_tw, Wp_cls_t, t);
        pack_tower_k<<<(t + 255) / 256, 256, 0, stream>>>(reg_tw, Wp_reg_t, t);
        int tc = 768 * 9 * 256;
        pack_out_k<<<(tc + 255) / 256, 256, 0, stream>>>(cls_wout, Wp_cls_o, 720, tc);
        int tr = 128 * 9 * 256;
        pack_out_k<<<(tr + 255) / 256, 256, 0, stream>>>(reg_wout, Wp_reg_o, 36, tr);
    }

    // 2) features NCHW f32 -> NHWC bf16
    for (int l = 0; l < 4; ++l) {
        int t = 4 * HWs[l] * 256;
        nchw2nhwc_k<<<(t + 255) / 256, 256, 0, stream>>>(feats[l], Xbuf + xoff[l], HWs[l]);
    }

    // 3) anchors
    {
        float* aout = (float*)d_out + (size_t)15667200 + 783360;
        anchors_k<<<(48960 + 255) / 256, 256, 0, stream>>>(img_h, img_w, aout);
    }

    // 4) both heads: 4 tower convs + output conv per level
    for (int hd = 0; hd < 2; ++hd) {
        unsigned short* Wt = hd ? Wp_reg_t : Wp_cls_t;
        unsigned short* Wo = hd ? Wp_reg_o : Wp_cls_o;
        const float* tb    = hd ? reg_tb : cls_tb;
        const float* bo    = hd ? reg_bout : cls_bout;
        const int CO_o     = hd ? 36 : 720;
        const int COpad_o  = hd ? 128 : 768;
        const int K        = hd ? 4 : 80;
        float* outp        = (float*)d_out + (hd ? (size_t)15667200 : 0);

        for (int l = 0; l < 4; ++l) {
            const int H = Hs[l], W = Hs[l];
            const int ptiles = 4 * HWs[l] / 64;
            unsigned short* cur = Xbuf + xoff[l];
            unsigned short* bufs[2] = {ping, pong};
            for (int t = 0; t < 4; ++t) {
                unsigned short* nxt = bufs[t & 1];
                conv3x3_wmma_k<<<dim3(2, ptiles), 256, 0, stream>>>(
                    cur, Wt + (size_t)t * 589824, tb + t * 256,
                    nxt, nullptr, H, W, 256, /*mode=*/0, 0, 1, 0, 0);
                cur = nxt;
            }
            conv3x3_wmma_k<<<dim3(COpad_o / 128, ptiles), 256, 0, stream>>>(
                cur, Wo, bo, nullptr, outp, H, W, CO_o, /*mode=*/1,
                9, K, lvl_base[l], NTOT);
        }
    }
}
